// MixAttention_47785806135939
// MI455X (gfx1250) — matrix operand order repaired and checked by reading, not compiled, not run on hardware
//
#include <hip/hip_runtime.h>

// ---------------------------------------------------------------------------
// MixAttention for MI455X (gfx1250, wave32, WMMA 16x16x32 f16, f32 accum)
// ---------------------------------------------------------------------------
// Shapes (fixed): B=4, L=2048, D=512, H=8, A=64. M = B*L = 8192.
// Pipeline:
//   1) 5x gemm_wmma : proj = X(f32) @ W(f32) + b  -> f16 in workspace
//   2) attn_kernel  : flash attention with dual QK^T, f16 WMMA, f32 online softmax
//   3) gemm_wmma    : out = ctx(f16) @ Wo + b + residual -> f32 tmp
//   4) ln_kernel    : LayerNorm(tmp) * g + b -> d_out (f32)
// ---------------------------------------------------------------------------

typedef __attribute__((ext_vector_type(16))) _Float16 v16h;
typedef __attribute__((ext_vector_type(8)))  float    v8f;

union AFrag { unsigned int u[8]; v16h v; };

__device__ __forceinline__ v8f wmma16x16x32(const AFrag& a, const AFrag& b, v8f c) {
  // 8 args: (neg_a, A, neg_b, B, c_mod, C, reuse_a, reuse_b)
  return __builtin_amdgcn_wmma_f32_16x16x32_f16(false, a.v, false, b.v,
                                                (short)0, c, false, false);
}

// ===========================================================================
// GEMM: out[M,512] = A[M,512] @ W[512,512] + bias (+ residual)
//   A is f16 (AF16=true) or f32; output f16 (F16OUT) or f32.
//   Block: 256 threads (8 waves). Block tile 128x128, wave tile 32x64,
//   K-step 32 via LDS. W is K-major [k][n]; transposed to [n][k] in LDS so
//   B-fragment pair loads along K are contiguous.
// ===========================================================================
template <bool AF16, bool F16OUT, bool ADD_RES>
__global__ __launch_bounds__(256) void gemm_wmma(
    const void* __restrict__ Aptr, const float* __restrict__ W,
    const float* __restrict__ bias, const float* __restrict__ residual,
    void* __restrict__ out) {
  constexpr int KTOT = 512;
  __shared__ _Float16 As[128][40];  // [m][k], pad 40 to dodge bank conflicts
  __shared__ _Float16 Bs[128][40];  // [n][k] (transposed W tile)

  const int bm  = blockIdx.x * 128;
  const int bn  = blockIdx.y * 128;
  const int t   = threadIdx.x;
  const int wid = t >> 5, lane = t & 31;
  const int wm  = (wid & 3) * 32;   // wave row offset  (4 waves down)
  const int wn  = (wid >> 2) * 64;  // wave col offset  (2 waves across)
  const int l15 = lane & 15;
  const int lhi = lane >> 4;        // 0 / 1
  const int kb  = lhi * 8;          // A-fragment K base per documented layout

  v8f acc[2][4] = {};

  for (int k0 = 0; k0 < KTOT; k0 += 32) {
    __syncthreads();
    // ---- load A tile (128 x 32) into LDS as f16 --------------------------
    if constexpr (AF16) {
      const _Float16* Ap = (const _Float16*)Aptr;
      for (int i = 0; i < 2; ++i) {
        int idx = t + i * 256, row = idx >> 2, seg = idx & 3;  // 8 f16 per seg
        uint4 d = *(const uint4*)(Ap + (size_t)(bm + row) * 512 + k0 + seg * 8);
        unsigned int* dst = (unsigned int*)&As[row][seg * 8];
        dst[0] = d.x; dst[1] = d.y; dst[2] = d.z; dst[3] = d.w;
      }
    } else {
      const float* Ap = (const float*)Aptr;
      for (int i = 0; i < 4; ++i) {
        int idx = t + i * 256, row = idx >> 3, seg = idx & 7;  // 4 f32 per seg
        float4 d = *(const float4*)(Ap + (size_t)(bm + row) * 512 + k0 + seg * 4);
        _Float16* dst = &As[row][seg * 4];
        dst[0] = (_Float16)d.x; dst[1] = (_Float16)d.y;
        dst[2] = (_Float16)d.z; dst[3] = (_Float16)d.w;
      }
    }
    // ---- load W tile (32 x 128), convert + transpose into Bs[n][k] -------
    for (int i = 0; i < 4; ++i) {
      int idx = t + i * 256, kk = idx >> 5, seg = idx & 31;    // 4 cols per seg
      float4 d = *(const float4*)(W + (size_t)(k0 + kk) * 512 + bn + seg * 4);
      Bs[seg * 4 + 0][kk] = (_Float16)d.x;
      Bs[seg * 4 + 1][kk] = (_Float16)d.y;
      Bs[seg * 4 + 2][kk] = (_Float16)d.z;
      Bs[seg * 4 + 3][kk] = (_Float16)d.w;
    }
    __syncthreads();

    // ---- prefetch next K tile (gfx1250 global_prefetch_b8) ---------------
    if (k0 + 32 < KTOT) {
      __builtin_prefetch(W + (size_t)(k0 + 32 + (t >> 3)) * 512 + bn + (t & 7) * 16, 0, 1);
      const char* ap = (const char*)Aptr;
      size_t esz = AF16 ? 2 : 4;
      __builtin_prefetch(ap + ((size_t)(bm + (t >> 1)) * 512 + (k0 + 32) + (t & 1) * 16) * esz, 0, 1);
    }

    // ---- build fragments and issue WMMAs ---------------------------------
    AFrag af[2], bf[4];
    for (int tm = 0; tm < 2; ++tm) {
      const unsigned int* src = (const unsigned int*)&As[wm + tm * 16 + l15][0];
      for (int j = 0; j < 4; ++j) af[tm].u[j]     = src[(kb + 2 * j) >> 1];
      for (int j = 0; j < 4; ++j) af[tm].u[4 + j] = src[(16 + kb + 2 * j) >> 1];
    }
    for (int tn = 0; tn < 4; ++tn) {
      const unsigned int* src = (const unsigned int*)&Bs[wn + tn * 16 + l15][0];
      const int base = lhi * 8;  // B-fragment K base
      for (int j = 0; j < 8; ++j) bf[tn].u[j] = src[(base >> 1) + (j & 3) + (j >> 2) * 8];
    }
    for (int tm = 0; tm < 2; ++tm)
      for (int tn = 0; tn < 4; ++tn)
        acc[tm][tn] = wmma16x16x32(af[tm], bf[tn], acc[tm][tn]);
  }

  // ---- epilogue: bias (+residual), store ---------------------------------
  for (int tm = 0; tm < 2; ++tm)
    for (int tn = 0; tn < 4; ++tn)
      for (int r = 0; r < 8; ++r) {
        const int row = bm + wm + tm * 16 + r + lhi * 8;  // C layout: M=r(+8)
        const int col = bn + wn + tn * 16 + l15;          // N = lane&15
        float v = acc[tm][tn][r] + bias[col];
        if constexpr (ADD_RES) v += residual[(size_t)row * 512 + col];
        if constexpr (F16OUT)
          ((_Float16*)out)[(size_t)row * 512 + col] = (_Float16)v;
        else
          ((float*)out)[(size_t)row * 512 + col] = v;
      }
}

// ===========================================================================
// Flash attention, dual QK^T summed before softmax.
//   grid = B * H * (L/128); 256 threads = 8 waves; wave owns 16 q-rows.
//   Q fragments (qd,qt) held in registers for the whole key loop.
//   Key blocks of 64 staged in LDS; vt transposed on write so P@V B-frags
//   read contiguous K-pairs; P round-trips LDS to become an A-fragment.
// ===========================================================================
__global__ __launch_bounds__(256) void attn_kernel(
    const _Float16* __restrict__ qd, const _Float16* __restrict__ qt,
    const _Float16* __restrict__ kd, const _Float16* __restrict__ kt,
    const _Float16* __restrict__ vt, _Float16* __restrict__ ctx) {
  constexpr int L = 2048, DM = 512;
  __shared__ _Float16 KD[64][72];     // [key][a]
  __shared__ _Float16 KT[64][72];     // [key][a]
  __shared__ _Float16 VT[64][72];     // [a][key]  (transposed)
  __shared__ _Float16 P[8][16][72];   // per-wave probs [row][key]

  const int bid = blockIdx.x;
  const int b   = bid >> 7;           // 8 heads * 16 q-blocks per batch
  const int rem = bid & 127;
  const int h   = rem >> 4;
  const int qb  = rem & 15;
  const int t = threadIdx.x, wid = t >> 5, lane = t & 31;
  const int l15 = lane & 15, lhi = lane >> 4;
  const int kb  = lhi * 8;
  const int rowb = b * L;
  const int hc   = h * 64;
  const int qrow = rowb + qb * 128 + wid * 16 + l15;

  // ---- load Q fragments (A-matrix layout, K = a) once --------------------
  AFrag qdf[2], qtf[2];
  for (int kc = 0; kc < 2; ++kc) {
    const unsigned int* sd = (const unsigned int*)(qd + (size_t)qrow * DM + hc + kc * 32);
    const unsigned int* st = (const unsigned int*)(qt + (size_t)qrow * DM + hc + kc * 32);
    for (int j = 0; j < 4; ++j) {
      qdf[kc].u[j]     = sd[(kb + 2 * j) >> 1];
      qtf[kc].u[j]     = st[(kb + 2 * j) >> 1];
      qdf[kc].u[4 + j] = sd[(16 + kb + 2 * j) >> 1];
      qtf[kc].u[4 + j] = st[(16 + kb + 2 * j) >> 1];
    }
  }

  v8f accO[4] = {};
  float mrow[8], lrow[8];
  for (int r = 0; r < 8; ++r) { mrow[r] = -1e30f; lrow[r] = 0.f; }
  const float scale = 0.125f;  // 1/sqrt(64)

  for (int k0 = 0; k0 < L; k0 += 64) {
    __syncthreads();
    // ---- stage 64-key tiles of kd, kt, vt (vt transposed) ---------------
    for (int i = 0; i < 2; ++i) {
      int idx = t + i * 256, row = idx >> 3, seg = idx & 7;  // 8 f16 per seg
      size_t g = (size_t)(rowb + k0 + row) * DM + hc + seg * 8;
      uint4 dd = *(const uint4*)(kd + g);
      uint4 dt = *(const uint4*)(kt + g);
      uint4 dv = *(const uint4*)(vt + g);
      unsigned int* p1 = (unsigned int*)&KD[row][seg * 8];
      p1[0] = dd.x; p1[1] = dd.y; p1[2] = dd.z; p1[3] = dd.w;
      unsigned int* p2 = (unsigned int*)&KT[row][seg * 8];
      p2[0] = dt.x; p2[1] = dt.y; p2[2] = dt.z; p2[3] = dt.w;
      union { uint4 q; _Float16 hh[8]; } tv; tv.q = dv;
      for (int c = 0; c < 8; ++c) VT[seg * 8 + c][row] = tv.hh[c];
    }
    __syncthreads();

    // ---- scores S = (qd*kd^T + qt*kt^T) ----------------------------------
    v8f s[4] = {};
    for (int tn = 0; tn < 4; ++tn) {
      const unsigned int* srcd = (const unsigned int*)&KD[tn * 16 + l15][0];
      const unsigned int* srct = (const unsigned int*)&KT[tn * 16 + l15][0];
      for (int kc = 0; kc < 2; ++kc) {
        AFrag bd, bt;
        const int base = kc * 32 + lhi * 8;
        for (int j = 0; j < 8; ++j) {
          bd.u[j] = srcd[(base >> 1) + (j & 3) + (j >> 2) * 8];
          bt.u[j] = srct[(base >> 1) + (j & 3) + (j >> 2) * 8];
        }
        s[tn] = wmma16x16x32(qdf[kc], bd, s[tn]);
        s[tn] = wmma16x16x32(qtf[kc], bt, s[tn]);
      }
    }

    // ---- online softmax (row = r + 8*lhi; reduce over 16-lane half) ------
    for (int r = 0; r < 8; ++r) {
      float v0 = s[0][r] * scale, v1 = s[1][r] * scale;
      float v2 = s[2][r] * scale, v3 = s[3][r] * scale;
      float v = fmaxf(fmaxf(v0, v1), fmaxf(v2, v3));
      v = fmaxf(v, __shfl_xor(v, 1, 32));
      v = fmaxf(v, __shfl_xor(v, 2, 32));
      v = fmaxf(v, __shfl_xor(v, 4, 32));
      v = fmaxf(v, __shfl_xor(v, 8, 32));
      const float mnew = fmaxf(mrow[r], v);
      const float corr = __expf(mrow[r] - mnew);
      mrow[r] = mnew;
      float p0 = __expf(v0 - mnew), p1 = __expf(v1 - mnew);
      float p2 = __expf(v2 - mnew), p3 = __expf(v3 - mnew);
      s[0][r] = p0; s[1][r] = p1; s[2][r] = p2; s[3][r] = p3;
      float ss = p0 + p1 + p2 + p3;
      ss += __shfl_xor(ss, 1, 32);
      ss += __shfl_xor(ss, 2, 32);
      ss += __shfl_xor(ss, 4, 32);
      ss += __shfl_xor(ss, 8, 32);
      lrow[r] = lrow[r] * corr + ss;
      for (int tn = 0; tn < 4; ++tn) accO[tn][r] *= corr;
    }

    // ---- spill P to LDS (C layout -> row-major f16) ----------------------
    for (int tn = 0; tn < 4; ++tn)
      for (int r = 0; r < 8; ++r)
        P[wid][r + lhi * 8][tn * 16 + l15] = (_Float16)s[tn][r];
    __syncthreads();

    // ---- O += P @ V ------------------------------------------------------
    for (int kc = 0; kc < 2; ++kc) {
      AFrag pf;
      const unsigned int* sp = (const unsigned int*)&P[wid][l15][0];
      for (int j = 0; j < 4; ++j) {
        pf.u[j]     = sp[(kc * 32 + kb + 2 * j) >> 1];
        pf.u[4 + j] = sp[(kc * 32 + 16 + kb + 2 * j) >> 1];
      }
      for (int tn = 0; tn < 4; ++tn) {
        AFrag vf;
        const unsigned int* sv = (const unsigned int*)&VT[tn * 16 + l15][0];
        const int base = kc * 32 + lhi * 8;
        for (int j = 0; j < 8; ++j) vf.u[j] = sv[(base >> 1) + (j & 3) + (j >> 2) * 8];
        accO[tn] = wmma16x16x32(pf, vf, accO[tn]);
      }
    }
  }

  // ---- normalize and store ctx (f16, [B*L, H*A] layout) ------------------
  for (int r = 0; r < 8; ++r) {
    const float inv = 1.0f / lrow[r];
    const int grow = rowb + qb * 128 + wid * 16 + r + lhi * 8;
    for (int tn = 0; tn < 4; ++tn)
      ctx[(size_t)grow * DM + hc + tn * 16 + l15] = (_Float16)(accO[tn][r] * inv);
  }
}

// ===========================================================================
// LayerNorm over D=512 per row. One block (256 threads) per row.
// ===========================================================================
__global__ __launch_bounds__(256) void ln_kernel(
    const float* __restrict__ x, const float* __restrict__ g,
    const float* __restrict__ bb, float* __restrict__ out) {
  __shared__ float ssum[8], ssq[8];
  __shared__ float mu, rstd;
  const int row = blockIdx.x, t = threadIdx.x;
  const float v0 = x[(size_t)row * 512 + t];
  const float v1 = x[(size_t)row * 512 + 256 + t];
  float s = v0 + v1, q = v0 * v0 + v1 * v1;
  for (int m = 1; m < 32; m <<= 1) {
    s += __shfl_xor(s, m, 32);
    q += __shfl_xor(q, m, 32);
  }
  const int wid = t >> 5, lane = t & 31;
  if (lane == 0) { ssum[wid] = s; ssq[wid] = q; }
  __syncthreads();
  if (t == 0) {
    float S = 0.f, Q = 0.f;
    for (int i = 0; i < 8; ++i) { S += ssum[i]; Q += ssq[i]; }
    const float m = S * (1.0f / 512.0f);
    const float var = Q * (1.0f / 512.0f) - m * m;
    mu = m;
    rstd = rsqrtf(var + 1e-5f);
  }
  __syncthreads();
  out[(size_t)row * 512 + t]       = (v0 - mu) * rstd * g[t] + bb[t];
  out[(size_t)row * 512 + 256 + t] = (v1 - mu) * rstd * g[256 + t] + bb[256 + t];
}

// ===========================================================================
// Host launcher
// ===========================================================================
extern "C" void kernel_launch(void* const* d_in, const int* in_sizes, int n_in,
                              void* d_out, int out_size, void* d_ws, size_t ws_size,
                              hipStream_t stream) {
  (void)in_sizes; (void)n_in; (void)out_size; (void)ws_size;
  constexpr int M = 4 * 2048;             // B*L
  constexpr size_t PROJ = (size_t)M * 512 * sizeof(_Float16);  // 8 MB each

  // inputs (setup_inputs dict order)
  const float* Xq_d = (const float*)d_in[0];
  const float* Xq_t = (const float*)d_in[1];
  const float* Xk_d = (const float*)d_in[2];
  const float* Xk_t = (const float*)d_in[3];
  const float* Xv_t = (const float*)d_in[4];
  const float* Wqd = (const float*)d_in[5];
  const float* Wqt = (const float*)d_in[6];
  const float* Wkd = (const float*)d_in[7];
  const float* Wkt = (const float*)d_in[8];
  const float* Wvt = (const float*)d_in[9];
  const float* Wo  = (const float*)d_in[10];
  const float* Bqd = (const float*)d_in[11];
  const float* Bqt = (const float*)d_in[12];
  const float* Bkd = (const float*)d_in[13];
  const float* Bkt = (const float*)d_in[14];
  const float* Bvt = (const float*)d_in[15];
  const float* Bo  = (const float*)d_in[16];
  const float* lng = (const float*)d_in[17];
  const float* lnb = (const float*)d_in[18];

  // workspace layout (64 MB total)
  char* ws = (char*)d_ws;
  _Float16* qd  = (_Float16*)(ws + 0 * PROJ);
  _Float16* qt  = (_Float16*)(ws + 1 * PROJ);
  _Float16* kd  = (_Float16*)(ws + 2 * PROJ);
  _Float16* kt  = (_Float16*)(ws + 3 * PROJ);
  _Float16* vt  = (_Float16*)(ws + 4 * PROJ);
  _Float16* ctx = (_Float16*)(ws + 5 * PROJ);
  float*    tmp = (float*)   (ws + 6 * PROJ);  // f32 pre-LN buffer (16 MB)

  const dim3 ggrid(M / 128, 512 / 128);
  const dim3 blk(256);

  // 1) projections: f32 A, f16 out
  gemm_wmma<false, true, false><<<ggrid, blk, 0, stream>>>(Xq_d, Wqd, Bqd, nullptr, qd);
  gemm_wmma<false, true, false><<<ggrid, blk, 0, stream>>>(Xq_t, Wqt, Bqt, nullptr, qt);
  gemm_wmma<false, true, false><<<ggrid, blk, 0, stream>>>(Xk_d, Wkd, Bkd, nullptr, kd);
  gemm_wmma<false, true, false><<<ggrid, blk, 0, stream>>>(Xk_t, Wkt, Bkt, nullptr, kt);
  gemm_wmma<false, true, false><<<ggrid, blk, 0, stream>>>(Xv_t, Wvt, Bvt, nullptr, vt);

  // 2) dual-QK flash attention: grid = B*H*(L/128) = 512
  attn_kernel<<<dim3(512), blk, 0, stream>>>(qd, qt, kd, kt, vt, ctx);

  // 3) output projection + residual: f16 A, f32 out
  gemm_wmma<true, false, true><<<ggrid, blk, 0, stream>>>(ctx, Wo, Bo, Xq_d, tmp);

  // 4) LayerNorm -> d_out
  ln_kernel<<<dim3(M), blk, 0, stream>>>(tmp, lng, lnb, (float*)d_out);
}
